// Operator_46050639347867
// MI455X (gfx1250) — matrix operand order repaired and checked by reading, not compiled, not run on hardware
//
#include <hip/hip_runtime.h>

// MI455X / gfx1250, wave32. Core contraction done with V_WMMA_F32_16X16X32_F16
// (f16 inputs, f32 accumulate): 8192 MACs/instruction vs 1024 for f32 WMMA.
// Problem is compute-bound (14.5 GFLOP on ~3MB of data, AI ~ 4800 FLOP/B), so
// the f16 matrix path is the right precision/throughput trade (rel err ~1e-3
// over the 3072-term contraction with exact f32 accumulation).

#define BB 4
#define PP 4
#define TT 16
#define CC 32
#define SS 96

typedef _Float16 v16h __attribute__((ext_vector_type(16)));
typedef _Float16 v8h  __attribute__((ext_vector_type(8)));
typedef float    v8f  __attribute__((ext_vector_type(8)));

// ---------------------------------------------------------------------------
// Kernel 1: precompute f16 operand panels (shared across all 256 (b,p,t)) and
// zero the g accumulator.
//   Kh   [s][alpha][c]  f16  : A-operand rows, c contiguous (K-dim contiguous)
//   Qfrag[s][n][g][j]   f16  : B-operand pre-swizzled to this code's contiguous-k B 32x16
//                              fragment layout: lane lo=n, half g (K=0..15 /
//                              16..31), 16 contiguous halfs per lane.
// ---------------------------------------------------------------------------
__global__ void prep_kernel(const float* __restrict__ Q,
                            const float* __restrict__ K,
                            _Float16* __restrict__ Kh,
                            _Float16* __restrict__ Qfrag,
                            float* __restrict__ g) {
    int i = blockIdx.x * 256 + threadIdx.x;
    if (i < BB * PP * SS) g[i] = 0.0f;            // 16*96 accumulators
    if (i < SS * SS * CC) {                        // 294912 elements
        // Kh[s*96*32 + a*32 + c] = K[a,s,c]
        int c = i & 31;
        int s = (i >> 5) % SS;
        int a = i / (SS * CC);
        Kh[s * (SS * CC) + a * CC + c] = (_Float16)K[a * (SS * CC) + s * CC + c];
        // Qfrag[((s*96+n)*2+gg)*16+j] = Q[n,s, gg*16+j]
        int j  = i & 15;
        int gg = (i >> 4) & 1;
        int n  = (i >> 5) % SS;
        int s2 = i / (SS * CC);
        Qfrag[i] = (_Float16)Q[n * (SS * CC) + s2 * CC + gg * 16 + j];
    }
}

// ---------------------------------------------------------------------------
// Kernel 2: one workgroup per (b,p,t). 384 threads = 12 waves. Wave w owns a
// (3 x 1) strip of 16x16 output tiles of the 96x96 kvt matrix:
//   beta column jcol = w % 6, alpha tiles 3*(w/6)+r, r=0..2.
// Loop s = 0..95: acc += A(Kh rows) x B(Qfrag col scaled by x[...,c,s]).
// Fragments for iteration s+1 are fetched BEFORE the s WMMAs issue (register
// double-buffering) so the s_wait_loadcnt ahead of each WMMA covers a full
// iteration of latency instead of a raw L2 round trip.
// Epilogue: fold Aoo[alpha,beta,t]*w[beta], reduce over beta, atomicAdd to g.
// ---------------------------------------------------------------------------
__global__ void kvt_wmma_kernel(const float* __restrict__ x,
                                const _Float16* __restrict__ Kh,
                                const _Float16* __restrict__ Qfrag,
                                const float* __restrict__ Aoo,
                                const float* __restrict__ w,
                                float* __restrict__ g) {
    __shared__ float xt[SS * CC];   // x slice transposed: xt[s*32 + c], 12 KB

    const int blk = blockIdx.x;            // (b*P + p)*T + t
    const int bp  = blk / TT;
    const int t   = blk % TT;

    // stage x[b,p,t,:,:] transposed into LDS
    const float* xg = x + blk * (CC * SS);
    for (int idx = threadIdx.x; idx < CC * SS; idx += blockDim.x) {
        int c = idx / SS, s = idx % SS;
        xt[s * CC + c] = xg[idx];
    }
    __syncthreads();

    const int wave  = threadIdx.x >> 5;
    const int lane  = threadIdx.x & 31;
    const int lo    = lane & 15;
    const int hi    = lane >> 4;           // 0: lanes 0-15, 1: lanes 16-31
    const int jcol  = wave % 6;            // beta tile column
    const int ahalf = wave / 6;            // 0 or 1 -> alpha tiles 0-2 / 3-5

    const int kbB  = hi ? 16 : 0;          // B fragment K offset per lane half
    const int kbA  = hi ? 16 : 0;          // A fragment K offset per lane half
    const int nIdx = jcol * 16 + lo;       // beta column owned by this lane
    const int aRow = ahalf * 48 + lo;      // base alpha row for this lane

    v8f acc[3];
#pragma unroll
    for (int r = 0; r < 3; ++r) acc[r] = (v8f){0.f,0.f,0.f,0.f,0.f,0.f,0.f,0.f};

    // ---- fragment fetch for one s ----------------------------------------
    auto load_frags = [&](int s, v16h& bbOut, v16h* aOut) {
        // B operand: Q[beta,s,c] * x[b,p,t,c,s], K-dim = c
        const _Float16* qb = Qfrag + (((s * SS + nIdx) << 1) + hi) * 16;
        v16h bq = *(const v16h*)qb;                 // 32B contiguous
        const float* xrow = xt + s * CC + kbB;      // 16 f32 (LDS broadcast)
        v16h bv;
#pragma unroll
        for (int j = 0; j < 16; ++j) bv[j] = (_Float16)xrow[j];
        bbOut = bq * bv;                            // v_pk_mul_f16
        // A operands: rows of Kh, contiguous K order (16 halfs per lane half)
#pragma unroll
        for (int r = 0; r < 3; ++r) {
            const _Float16* ka = Kh + (s * SS + aRow + r * 16) * CC + kbA;
            v8h a0 = *(const v8h*)ka;               // K = kbA .. kbA+7
            v8h a1 = *(const v8h*)(ka + 8);         // K = kbA+8 .. kbA+15
            aOut[r] = __builtin_shufflevector(a0, a1, 0,1,2,3,4,5,6,7,
                                                      8,9,10,11,12,13,14,15);
        }
    };

    v16h bb_cur, a_cur[3];
    load_frags(0, bb_cur, a_cur);

    for (int s = 0; s < SS; ++s) {
        if (s + 8 < SS) {
            __builtin_prefetch(Kh + (s + 8) * (SS * CC), 0, 1);
            __builtin_prefetch(Qfrag + (s + 8) * (SS * CC), 0, 1);
        }
        v16h bb_nxt, a_nxt[3];
        if (s + 1 < SS) load_frags(s + 1, bb_nxt, a_nxt);  // issue early

#pragma unroll
        for (int r = 0; r < 3; ++r) {
            acc[r] = __builtin_amdgcn_wmma_f32_16x16x32_f16(
                false, a_cur[r], false, bb_cur, (short)0, acc[r], false, false);
        }
        bb_cur = bb_nxt;
#pragma unroll
        for (int r = 0; r < 3; ++r) a_cur[r] = a_nxt[r];
    }

    // ---- epilogue: g[bp,alpha] += sum_beta Aoo[alpha,beta,t]*w[beta]*kvt --
    float* gbp = g + bp * SS;
    const float wbeta = w[nIdx];
#pragma unroll
    for (int r2 = 0; r2 < 3; ++r2) {
#pragma unroll
        for (int r = 0; r < 8; ++r) {
            int alpha = ahalf * 48 + r2 * 16 + r + (hi ? 8 : 0);
            float v = acc[r2][r] * Aoo[(alpha * SS + nIdx) * TT + t] * wbeta;
            v += __shfl_xor(v, 1, 32);
            v += __shfl_xor(v, 2, 32);
            v += __shfl_xor(v, 4, 32);
            v += __shfl_xor(v, 8, 32);
            if (lo == 0) atomicAdd(&gbp[alpha], v);   // alpha uniform per half
        }
    }
}

// ---------------------------------------------------------------------------
// Kernel 3: y[b,p,c] = sum_s x[b,p,T-1,c,s] * (sum_alpha V[alpha,s,c]*g[bp,alpha])
// 16 blocks (one per (b,p)), tiny FLOP count -> plain VALU.
// ---------------------------------------------------------------------------
__global__ void finalize_kernel(const float* __restrict__ x,
                                const float* __restrict__ V,
                                const float* __restrict__ g,
                                float* __restrict__ out) {
    __shared__ float gl[SS];
    __shared__ float xl[CC * SS];
    __shared__ float yacc[CC];
    const int bp  = blockIdx.x;
    const int tid = threadIdx.x;

    if (tid < SS) gl[tid] = g[bp * SS + tid];
    if (tid < CC) yacc[tid] = 0.0f;
    const float* xlast = x + (bp * TT + (TT - 1)) * (CC * SS);
    for (int i = tid; i < CC * SS; i += 256) xl[i] = xlast[i];
    __syncthreads();

    for (int idx = tid; idx < SS * CC; idx += 256) {
        int s = idx / CC, c = idx % CC;
        float u = 0.0f;
#pragma unroll 4
        for (int a = 0; a < SS; ++a) u += V[a * (SS * CC) + s * CC + c] * gl[a];
        atomicAdd(&yacc[c], u * xl[c * SS + s]);
    }
    __syncthreads();
    if (tid < CC) out[bp * CC + tid] = yacc[tid];
}

// ---------------------------------------------------------------------------
extern "C" void kernel_launch(void* const* d_in, const int* in_sizes, int n_in,
                              void* d_out, int out_size, void* d_ws, size_t ws_size,
                              hipStream_t stream) {
    const float* x   = (const float*)d_in[0];
    const float* Q   = (const float*)d_in[1];
    const float* K   = (const float*)d_in[2];
    const float* V   = (const float*)d_in[3];
    const float* Aoo = (const float*)d_in[4];
    const float* w   = (const float*)d_in[5];

    // workspace carve-up (~1.19 MB): Kh | Qfrag | g
    char* ws = (char*)d_ws;
    const size_t panelBytes = (size_t)SS * SS * CC * sizeof(_Float16); // 589824
    _Float16* Kh    = (_Float16*)ws;
    _Float16* Qfrag = (_Float16*)(ws + panelBytes);
    float*    g     = (float*)(ws + 2 * panelBytes);

    prep_kernel<<<(SS * SS * CC + 255) / 256, 256, 0, stream>>>(Q, K, Kh, Qfrag, g);
    kvt_wmma_kernel<<<BB * PP * TT, 384, 0, stream>>>(x, Kh, Qfrag, Aoo, w, g);
    finalize_kernel<<<BB * PP, 256, 0, stream>>>(x, V, g, (float*)d_out);
}
